// Decoder_75196287418458
// MI455X (gfx1250) — matrix operand order repaired and checked by reading, not compiled, not run on hardware
//
#include <hip/hip_runtime.h>
#include <math.h>

// ---------------------------------------------------------------------------
// CDNA5 (gfx1250) WMMA implementation of the progressive 3D decoder.
//  - all channel contractions on v_wmma_f32_16x16x32_bf16 (wave32)
//  - weights pre-packed to bf16 in B-fragment order (coalesced b128 loads,
//    single running pointer per K-chunk)
//  - NCT accumulator tiles per wave (16 voxels x 16*NCT couts), 4 waves/block
//  - transpose-conv A tiles fetched by the Tensor Data Mover into a
//    double-buffered LDS tile: issue chunk k+1, s_wait_tensorcnt 1, compute
//    chunk k (DMA overlapped with the matrix pipe via the in-order TENSORcnt)
// ---------------------------------------------------------------------------

typedef __bf16 bf16;
typedef bf16  v16bf __attribute__((ext_vector_type(16)));
typedef bf16  v8bf  __attribute__((ext_vector_type(8)));
typedef float v8f   __attribute__((ext_vector_type(8)));
typedef float v4f   __attribute__((ext_vector_type(4)));
typedef unsigned int v4u __attribute__((ext_vector_type(4)));
typedef int   v8i   __attribute__((ext_vector_type(8)));
typedef int   v4i   __attribute__((ext_vector_type(4)));

#ifndef __has_builtin
#define __has_builtin(x) 0
#endif

#if __has_builtin(__builtin_amdgcn_tensor_load_to_lds)
#define HAVE_TDM 1
#else
#define HAVE_TDM 0
#endif

#define LDA_BF 40   // bf16 LDS row stride for 16x32 A tile (80B: 16B-aligned rows)

__device__ __forceinline__ float elu1(float v) {
    return v > 0.f ? v : expm1f(v);
}

__device__ __forceinline__ void wait_tensorcnt0() {
#if __has_builtin(__builtin_amdgcn_s_wait_tensorcnt)
    __builtin_amdgcn_s_wait_tensorcnt(0);
#else
    asm volatile("s_wait_tensorcnt 0x0" ::: "memory");
#endif
}
__device__ __forceinline__ void wait_tensorcnt1() {
#if __has_builtin(__builtin_amdgcn_s_wait_tensorcnt)
    __builtin_amdgcn_s_wait_tensorcnt(1);
#else
    asm volatile("s_wait_tensorcnt 0x1" ::: "memory");
#endif
}

// Issue (no wait) a TDM 2D tile load: 16 rows x 32 fp32, row stride =
// strideElems; rows beyond rowsValid zero-filled by the tensor OOB rule.
__device__ __forceinline__ void tdm_issue_tile_f32(float* ldsDst, const float* gsrc,
                                                   int rowsValid, int strideElems) {
#if HAVE_TDM
    unsigned lds = (unsigned)(uintptr_t)ldsDst;
    unsigned long long ga = (unsigned long long)(uintptr_t)gsrc;
    v4u g0;
    g0[0] = 1u;                                   // count=1 (valid descriptor)
    g0[1] = lds;                                  // lds_addr
    g0[2] = (unsigned)ga;                         // global_addr[31:0]
    g0[3] = (unsigned)((ga >> 32) & 0x01ffffffu) | (2u << 30); // addr[56:32] | type=2
    v8i g1;
    g1[0] = (2 << 16);                            // data_size = 4B
    g1[1] = (32 & 0xffff) << 16;                  // tensor_dim0 lo16 (= tile width)
    g1[2] = ((32 >> 16) & 0xffff) | ((rowsValid & 0xffff) << 16); // dim0 hi | dim1 lo
    g1[3] = (((unsigned)rowsValid >> 16) & 0xffff) | (32 << 16);  // dim1 hi | tile_dim0=32
    g1[4] = 16;                                   // tile_dim1=16, tile_dim2=0
    g1[5] = strideElems;                          // tensor_dim0_stride lo32
    g1[6] = 0;
    g1[7] = 0;
    v4i z = {0, 0, 0, 0};
#if defined(__clang_major__) && __clang_major__ >= 23
    v8i z8 = {};
    __builtin_amdgcn_tensor_load_to_lds(g0, g1, z, z, z8, 0);
#else
    __builtin_amdgcn_tensor_load_to_lds(g0, g1, z, z, 0);
#endif
#else
    (void)ldsDst; (void)gsrc; (void)rowsValid; (void)strideElems;
#endif
}

// A fragment (16x32, contiguous k = 16*(lane>>4) + s, as packed B) from an fp32 LDS tile (stride 32).
__device__ __forceinline__ v16bf a_frag_from_f32(const float* As32, int lane) {
    int m  = lane & 15;
    int hi = (lane >> 4) << 4;                     // 0 or 16
    const float* row = As32 + m * 32;
    v4f p0 = *(const v4f*)(row + hi);
    v4f p1 = *(const v4f*)(row + hi + 4);
    v4f p2 = *(const v4f*)(row + hi + 8);
    v4f p3 = *(const v4f*)(row + hi + 12);
    v16bf a;
#pragma unroll
    for (int i = 0; i < 4; ++i) {
        a[i]      = (bf16)p0[i];
        a[4 + i]  = (bf16)p1[i];
        a[8 + i]  = (bf16)p2[i];
        a[12 + i] = (bf16)p3[i];
    }
    return a;
}

// A fragment from a bf16 LDS tile (stride LDA_BF), vectorized ds_load_b128.
__device__ __forceinline__ v16bf a_frag_from_bf16(const bf16* As, int lane) {
    int m  = lane & 15;
    int hi = (lane >> 4) << 4;
    const bf16* row = As + m * LDA_BF;
    v8bf p0 = *(const v8bf*)(row + hi);
    v8bf p1 = *(const v8bf*)(row + 8 + hi);
    v16bf a;
#pragma unroll
    for (int i = 0; i < 8; ++i) { a[i] = p0[i]; a[8 + i] = p1[i]; }
    return a;
}

// ---------------------------------------------------------------------------
// Weight packing: fp32 [ntaps][Cin][Cout] -> bf16 B-fragments.
// Flat order: (((t*nCh + ck)*nCT + ct)*32 + lane)*16 ; lane holds 16 contiguous
// bf16 (32B) = its K-half (lane>=16 -> K 16..31) at column n = ct*16 + (lane&15).
// ---------------------------------------------------------------------------
__global__ void pack_weights_kernel(const float* __restrict__ W,
                                    bf16* __restrict__ P,
                                    int Cin, int Cout, int ntaps) {
    int gid = blockIdx.x * blockDim.x + threadIdx.x;
    int nCh = (Cin + 31) >> 5;
    int nCT = Cout >> 4;
    int total = ntaps * nCh * nCT * 32;
    if (gid >= total) return;
    int l = gid & 31;
    int r = gid >> 5;
    int ct = r % nCT; r /= nCT;
    int ck = r % nCh;
    int t  = r / nCh;
    int n  = ct * 16 + (l & 15);
    int kb = (l >> 4) << 4;
    v16bf frag;
#pragma unroll
    for (int i = 0; i < 16; ++i) {
        int ci = ck * 32 + kb + i;
        frag[i] = (ci < Cin) ? (bf16)W[((size_t)t * Cin + ci) * Cout + n] : (bf16)0.f;
    }
    *(v16bf*)(P + (size_t)gid * 16) = frag;
}

// ---------------------------------------------------------------------------
// Transpose conv k2 s2 + BN + *mask + ELU. Block = 128 threads (4 waves),
// each wave owns NCT 16x16 tiles. A tiles double-buffered in LDS via the TDM.
// grid = (parentTiles, ceil(nCT/(4*NCT)), 8 child parities).
// ---------------------------------------------------------------------------
template <int NCT>
__global__ void tconv_wmma_kernel(const float* __restrict__ X,
                                  const bf16* __restrict__ P,
                                  const float* __restrict__ bn_g,
                                  const float* __restrict__ bn_b,
                                  const float* __restrict__ bn_m,
                                  const float* __restrict__ bn_v,
                                  const unsigned char* __restrict__ maskIn,
                                  unsigned char* __restrict__ maskOut,
                                  float* __restrict__ Y,
                                  int Dp, int lDp, int Cin, int Cout)
{
    const int tid  = threadIdx.x;
    const int lane = tid & 31;
    const int wave = tid >> 5;
    const int tile = blockIdx.x;
    const int off  = blockIdx.z;
    const int oz = (off >> 2) & 1, oy = (off >> 1) & 1, ox = off & 1;
    const int nvox = Dp * Dp * Dp;
    const int nCh  = Cin >> 5;          // Cin is a multiple of 32 here
    const int nCT  = Cout >> 4;
    const int gct0 = (blockIdx.y * 4 + wave) * NCT;
    const bool active = gct0 < nCT;     // wave-uniform

    __shared__ __align__(16) float As32[2][16 * 32];

    v8f acc[NCT];
#pragma unroll
    for (int c = 0; c < NCT; ++c) acc[c] = (v8f){};

    const int rowsValid = nvox - tile * 16;
    const float* __restrict__ Xt = X + (size_t)tile * 16 * Cin;

    // Running packed-B pointer: layout walks (off*nCh + ck) linearly.
    const bf16* __restrict__ bp =
        P + ((size_t)(off * nCh) * nCT + gct0) * 512 + (size_t)lane * 16;
    const size_t bstep = (size_t)nCT * 512;

#if HAVE_TDM
    // Software pipeline: DMA chunk k+1 while the matrix pipe eats chunk k.
    if (wave == 0)
        tdm_issue_tile_f32(&As32[0][0], Xt, rowsValid, Cin);
    for (int ck = 0; ck < nCh; ++ck) {
        const int cur = ck & 1;
        if (wave == 0) {
            if (ck + 1 < nCh) {
                tdm_issue_tile_f32(&As32[cur ^ 1][0], Xt + (size_t)(ck + 1) * 32,
                                   rowsValid, Cin);
                wait_tensorcnt1();      // chunk ck landed; ck+1 still in flight
            } else {
                wait_tensorcnt0();
            }
        }
        __syncthreads();
        if (active) {
            v16bf a = a_frag_from_f32(&As32[cur][0], lane);
#pragma unroll
            for (int ct = 0; ct < NCT; ++ct) {
                const v16bf b = *(const v16bf*)(bp + ct * 512);
                acc[ct] = __builtin_amdgcn_wmma_f32_16x16x32_bf16(
                    false, a, false, b, (short)0, acc[ct], false, false);
            }
        }
        bp += bstep;
        __syncthreads();                // reads done before next overwrite
    }
#else
    for (int ck = 0; ck < nCh; ++ck) {
        __syncthreads();
#pragma unroll
        for (int j = 0; j < 4; ++j) {
            int row = wave + 4 * j;
            int pv  = tile * 16 + row;
            As32[0][row * 32 + lane] =
                (pv < nvox) ? Xt[(size_t)pv * Cin - (size_t)tile * 16 * Cin +
                                 (size_t)pv * 0 + (size_t)row * Cin + ck * 32 + lane]
                            : 0.f;
        }
        __syncthreads();
        if (active) {
            v16bf a = a_frag_from_f32(&As32[0][0], lane);
#pragma unroll
            for (int ct = 0; ct < NCT; ++ct) {
                const v16bf b = *(const v16bf*)(bp + ct * 512);
                acc[ct] = __builtin_amdgcn_wmma_f32_16x16x32_bf16(
                    false, a, false, b, (short)0, acc[ct], false, false);
            }
        }
        bp += bstep;
    }
#endif

    if (!active) return;

    const int n  = lane & 15;
    const int hi = (lane >> 4) << 3;
    const int Dc = Dp * 2;
    const int Dm = Dp - 1;

#pragma unroll
    for (int ct = 0; ct < NCT; ++ct) {
        int co = (gct0 + ct) * 16 + n;
        float scale = bn_g[co] * rsqrtf(bn_v[co] + 1e-5f);
        float mm = bn_m[co], bb = bn_b[co];
#pragma unroll
        for (int r = 0; r < 8; ++r) {
            int pv = tile * 16 + r + hi;
            if (pv >= nvox) continue;
            int px = pv & Dm, py = (pv >> lDp) & Dm, pz = pv >> (2 * lDp);
            size_t cv = ((size_t)(2 * pz + oz) * Dc + (2 * py + oy)) * Dc + (2 * px + ox);
            unsigned char pm = maskIn ? maskIn[pv] : (unsigned char)1;
            float mk  = pm ? 1.f : 0.f;
            float val = (acc[ct][r] - mm) * scale + bb;   // BN
            val = elu1(val * mk);                         // *mask, ELU
            Y[cv * Cout + co] = val;
            if (co == 0 && maskOut) maskOut[cv] = pm;
        }
    }
}

// ---------------------------------------------------------------------------
// Conv3d k3 SAME (implicit GEMM, 27 taps) + *mask + BN + *mask + ELU.
// Boundary-aware manual LDS staging; per-thread staging coordinates hoisted
// out of the tap/chunk loops (only the neighbor offset changes per tap).
// ---------------------------------------------------------------------------
template <int NCT>
__global__ void conv3_wmma_kernel(const float* __restrict__ X,
                                  const bf16* __restrict__ P,
                                  const float* __restrict__ bn_g,
                                  const float* __restrict__ bn_b,
                                  const float* __restrict__ bn_m,
                                  const float* __restrict__ bn_v,
                                  const unsigned char* __restrict__ mask,
                                  float* __restrict__ Y,
                                  int D, int lD, int C)
{
    const int tid  = threadIdx.x;
    const int lane = tid & 31;
    const int wave = tid >> 5;
    const int tile = blockIdx.x;
    const int nvox = D * D * D;
    const int nCh  = (C + 31) >> 5;
    const int nCT  = C >> 4;
    const int gct0 = (blockIdx.y * 4 + wave) * NCT;
    const bool active = gct0 < nCT;     // wave-uniform
    const int Dm = D - 1;

    __shared__ __align__(16) bf16 As[16 * LDA_BF];

    v8f acc[NCT];
#pragma unroll
    for (int c = 0; c < NCT; ++c) acc[c] = (v8f){};

    // Hoisted per-thread staging coordinates: this thread stages rows wave+4j,
    // column `lane` of the 16x32 A tile.
    int vx[4], vy[4], vz[4];
    bool vok[4];
#pragma unroll
    for (int j = 0; j < 4; ++j) {
        int ov = tile * 16 + wave + 4 * j;
        vok[j] = ov < nvox;
        vx[j] = ov & Dm;
        vy[j] = (ov >> lD) & Dm;
        vz[j] = ov >> (2 * lD);
    }
    const bool ciOk = lane < C;         // only false for C=16, lanes 16..31

    const bf16* __restrict__ bp = P + (size_t)gct0 * 512 + (size_t)lane * 16;
    const size_t bstep = (size_t)nCT * 512;

    for (int t = 0; t < 27; ++t) {
        const int dz = t / 9 - 1;
        const int dy = (t / 3) % 3 - 1;
        const int dx = t % 3 - 1;

        // Per-tap neighbor base offsets (loop-invariant over ck).
        size_t nbase[4];
        bool nok[4];
#pragma unroll
        for (int j = 0; j < 4; ++j) {
            int nx = vx[j] + dx, ny = vy[j] + dy, nz = vz[j] + dz;
            nok[j] = vok[j] && (unsigned)nx < (unsigned)D &&
                     (unsigned)ny < (unsigned)D && (unsigned)nz < (unsigned)D;
            nbase[j] = (((size_t)nz << (2 * lD)) + ((size_t)ny << lD) + (size_t)nx) * C;
        }

        for (int ck = 0; ck < nCh; ++ck) {
            const int ci = ck * 32 + lane;
            __syncthreads();
#pragma unroll
            for (int j = 0; j < 4; ++j) {
                float val = (nok[j] && ciOk) ? X[nbase[j] + ci] : 0.f;
                As[(wave + 4 * j) * LDA_BF + lane] = (bf16)val;
            }
            __syncthreads();

            if (active) {
                v16bf a = a_frag_from_bf16(As, lane);
                __builtin_prefetch(bp + bstep, 0, 1);
#pragma unroll
                for (int ct = 0; ct < NCT; ++ct) {
                    const v16bf b = *(const v16bf*)(bp + ct * 512);
                    acc[ct] = __builtin_amdgcn_wmma_f32_16x16x32_bf16(
                        false, a, false, b, (short)0, acc[ct], false, false);
                }
            }
            bp += bstep;
        }
    }

    if (!active) return;

    const int n  = lane & 15;
    const int hi = (lane >> 4) << 3;

#pragma unroll
    for (int ct = 0; ct < NCT; ++ct) {
        int co = (gct0 + ct) * 16 + n;
        float scale = bn_g[co] * rsqrtf(bn_v[co] + 1e-5f);
        float mm = bn_m[co], bb = bn_b[co];
#pragma unroll
        for (int r = 0; r < 8; ++r) {
            int ov = tile * 16 + r + hi;
            if (ov >= nvox) continue;
            float mk  = mask[ov] ? 1.f : 0.f;
            float val = acc[ct][r] * mk;            // conv * mask
            val = (val - mm) * scale + bb;          // BN
            val = elu1(val * mk);                   // *mask, ELU
            Y[(size_t)ov * C + co] = val;
        }
    }
}

// ---------------------------------------------------------------------------
// Per-level classifier / target / prune (bandwidth-trivial).
// ---------------------------------------------------------------------------
__global__ void classifier_kernel(float* __restrict__ X,
                                  const float* __restrict__ w,
                                  const float* __restrict__ bias,
                                  unsigned char* __restrict__ mask,
                                  const unsigned char* __restrict__ occ,
                                  float* __restrict__ logits_out,
                                  float* __restrict__ target_out,
                                  float* __restrict__ x_final_out,
                                  float* __restrict__ mask_final_out,
                                  int D, int lD, int C, int f, int RESr)
{
    int v = blockIdx.x * blockDim.x + threadIdx.x;
    int nvox = D * D * D;
    if (v >= nvox) return;
    int Dm = D - 1;

    float acc = bias[0];
    for (int c = 0; c < C; ++c) acc += X[(size_t)v * C + c] * w[c];

    unsigned char m0 = mask[v];
    float mk = m0 ? 1.f : 0.f;
    float logit = acc * mk;
    logits_out[v] = logit;

    int x = v & Dm, y = (v >> lD) & Dm, z = v >> (2 * lD);
    int any = 0;
    for (int dz = 0; dz < f && !any; ++dz)
        for (int dy = 0; dy < f && !any; ++dy)
            for (int dx = 0; dx < f; ++dx) {
                int zz = z * f + dz, yy = y * f + dy, xx = x * f + dx;
                if (occ[((size_t)zz * RESr + yy) * RESr + xx]) { any = 1; break; }
            }
    target_out[v] = (any && m0) ? 1.f : 0.f;

    int keep = (logit > 0.f) && m0;
    mask[v] = (unsigned char)keep;
    float km = keep ? 1.f : 0.f;
    for (int c = 0; c < C; ++c) {
        float val = X[(size_t)v * C + c] * km;
        X[(size_t)v * C + c] = val;
        if (x_final_out) x_final_out[(size_t)v * C + c] = val;
    }
    if (mask_final_out) mask_final_out[v] = km;
}

// ---------------------------------------------------------------------------
// Host-side launch helpers (template dispatch on NCT)
// ---------------------------------------------------------------------------
struct BNp { const float *b, *g, *m, *v; };

static inline int ilog2(int x) { int l = 0; while ((1 << l) < x) ++l; return l; }

static inline void pick_nct(int nCT, int& NCT, int& gridY) {
    NCT = (nCT >= 16) ? 4 : (nCT >= 8 ? 2 : 1);
    gridY = (nCT + 4 * NCT - 1) / (4 * NCT);
}

static void launch_pack(const float* W, bf16* P, int Cin, int Cout, int ntaps,
                        hipStream_t s) {
    int nCh = (Cin + 31) >> 5;
    int total = ntaps * nCh * (Cout >> 4) * 32;
    pack_weights_kernel<<<(total + 255) / 256, 256, 0, s>>>(W, P, Cin, Cout, ntaps);
}

static void launch_tconv(const float* X, const bf16* P, BNp bn,
                         const unsigned char* mIn, unsigned char* mOut, float* Y,
                         int Dp, int Cin, int Cout, hipStream_t s) {
    int nvox = Dp * Dp * Dp;
    int NCT, gy; pick_nct(Cout >> 4, NCT, gy);
    int lDp = ilog2(Dp);
    dim3 grid((nvox + 15) / 16, gy, 8);
    switch (NCT) {
    case 4: tconv_wmma_kernel<4><<<grid, 128, 0, s>>>(X, P, bn.g, bn.b, bn.m, bn.v,
                                                      mIn, mOut, Y, Dp, lDp, Cin, Cout); break;
    case 2: tconv_wmma_kernel<2><<<grid, 128, 0, s>>>(X, P, bn.g, bn.b, bn.m, bn.v,
                                                      mIn, mOut, Y, Dp, lDp, Cin, Cout); break;
    default: tconv_wmma_kernel<1><<<grid, 128, 0, s>>>(X, P, bn.g, bn.b, bn.m, bn.v,
                                                       mIn, mOut, Y, Dp, lDp, Cin, Cout); break;
    }
}

static void launch_conv3(const float* X, const bf16* P, BNp bn,
                         const unsigned char* M, float* Y, int D, int C,
                         hipStream_t s) {
    int NCT, gy; pick_nct(C >> 4, NCT, gy);
    int lD = ilog2(D);
    dim3 grid((D * D * D + 15) / 16, gy);
    switch (NCT) {
    case 4: conv3_wmma_kernel<4><<<grid, 128, 0, s>>>(X, P, bn.g, bn.b, bn.m, bn.v,
                                                      M, Y, D, lD, C); break;
    case 2: conv3_wmma_kernel<2><<<grid, 128, 0, s>>>(X, P, bn.g, bn.b, bn.m, bn.v,
                                                      M, Y, D, lD, C); break;
    default: conv3_wmma_kernel<1><<<grid, 128, 0, s>>>(X, P, bn.g, bn.b, bn.m, bn.v,
                                                       M, Y, D, lD, C); break;
    }
}

// ---------------------------------------------------------------------------
// Host orchestration
// ---------------------------------------------------------------------------
extern "C" void kernel_launch(void* const* d_in, const int* in_sizes, int n_in,
                              void* d_out, int out_size, void* d_ws, size_t ws_size,
                              hipStream_t stream) {
    (void)in_sizes; (void)out_size; (void)ws_size;

    int idx = 1;  // d_in[0] = z_feat; params leaves follow in sorted pytree order
    auto NXT = [&]() { return (const float*)d_in[idx++]; };
    auto NBN = [&]() { BNp r; r.b = NXT(); r.g = NXT(); r.m = NXT(); r.v = NXT(); return r; };

    // b1: bn1..bn4, c1w, c2w, t1w, t2w (keys sorted)
    BNp b1_bn1 = NBN(), b1_bn2 = NBN(), b1_bn3 = NBN(), b1_bn4 = NBN();
    const float* c1w = NXT(); const float* c2w = NXT();
    const float* t1w = NXT(); const float* t2w = NXT();
    // b2..b6: bn1, bn2, cw, tw
    BNp bn1_[7], bn2_[7];
    const float* cw_[7]; const float* tw_[7];
    for (int i = 2; i <= 6; ++i) { bn1_[i] = NBN(); bn2_[i] = NBN(); cw_[i] = NXT(); tw_[i] = NXT(); }
    // cls: bb1..bb6, w1..w6
    const float* bb_[7]; const float* w_[7];
    for (int i = 1; i <= 6; ++i) bb_[i] = NXT();
    for (int i = 1; i <= 6; ++i) w_[i]  = NXT();

    const float* z_feat = (const float*)d_in[0];
    const unsigned char* occ = (const unsigned char*)d_in[n_in - 1];

    static const int CH[7] = {1024, 512, 256, 128, 64, 32, 16};
    const int RESr = 128;

    // Workspace: two ping-pong activation buffers + masks + packed-weight slab.
    const size_t XBYTES = (size_t)128 * 128 * 128 * 16 * sizeof(float);
    const size_t MBYTES = (size_t)128 * 128 * 128;
    char* ws = (char*)d_ws;
    float* Xa = (float*)ws;
    float* Xb = (float*)(ws + XBYTES);
    unsigned char* Ma = (unsigned char*)(ws + 2 * XBYTES);
    unsigned char* Mb = Ma + MBYTES;
    bf16* Wpack = (bf16*)(ws + 2 * XBYTES + 2 * MBYTES);   // <= 57MB (c1w)

    // Output layout: out_cls[1..6], targets[1..6], x(128^3*16), mask(128^3)
    float* out = (float*)d_out;
    size_t clsSize[7];
    for (int i = 1; i <= 6; ++i) { int D = 4 << (i - 1); clsSize[i] = (size_t)D * D * D; }
    float* logitsPtr[7]; float* targetPtr[7];
    size_t o = 0;
    for (int i = 1; i <= 6; ++i) { logitsPtr[i] = out + o; o += clsSize[i]; }
    for (int i = 1; i <= 6; ++i) { targetPtr[i] = out + o; o += clsSize[i]; }
    float* xOut    = out + o; o += (size_t)128 * 128 * 128 * 16;
    float* maskOut = out + o;

    auto classify = [&](float* X, int i, unsigned char* M, int D, int C,
                        float* xf, float* mf) {
        int nvox = D * D * D;
        classifier_kernel<<<(nvox + 127) / 128, 128, 0, stream>>>(
            X, w_[i], bb_[i], M, occ, logitsPtr[i], targetPtr[i],
            xf, mf, D, ilog2(D), C, RESr / D, RESr);
    };

    // Block b1: 1 -> 2 -> 4 resolution
    launch_pack(t1w, Wpack, CH[0], CH[0], 8, stream);
    launch_tconv(z_feat, Wpack, b1_bn1, nullptr, Ma, Xa, 1, CH[0], CH[0], stream);
    launch_pack(c1w, Wpack, CH[0], CH[0], 27, stream);
    launch_conv3(Xa, Wpack, b1_bn2, Ma, Xb, 2, CH[0], stream);
    launch_pack(t2w, Wpack, CH[0], CH[1], 8, stream);
    launch_tconv(Xb, Wpack, b1_bn3, Ma, Mb, Xa, 2, CH[0], CH[1], stream);
    launch_pack(c2w, Wpack, CH[1], CH[1], 27, stream);
    launch_conv3(Xa, Wpack, b1_bn4, Mb, Xb, 4, CH[1], stream);

    classify(Xb, 1, Mb, 4, CH[1], nullptr, nullptr);

    // Levels 2..6
    int D = 4, Cprev = CH[1];
    float* Xcur = Xb; float* Xalt = Xa;
    unsigned char* Mcur = Mb; unsigned char* Mnext = Ma;
    for (int i = 2; i <= 6; ++i) {
        int C = CH[i];
        launch_pack(tw_[i], Wpack, Cprev, C, 8, stream);
        launch_tconv(Xcur, Wpack, bn1_[i], Mcur, Mnext, Xalt, D, Cprev, C, stream);
        D *= 2;
        launch_pack(cw_[i], Wpack, C, C, 27, stream);
        launch_conv3(Xalt, Wpack, bn2_[i], Mnext, Xcur, D, C, stream);
        classify(Xcur, i, Mnext, D, C,
                 (i == 6) ? xOut : nullptr, (i == 6) ? maskOut : nullptr);
        unsigned char* tmp = Mcur; Mcur = Mnext; Mnext = tmp;
        Cprev = C;
    }
}
